// PointNeXtSetAbstraction_49503793054298
// MI455X (gfx1250) — matrix operand order repaired and checked by reading, not compiled, not run on hardware
//
#include <hip/hip_runtime.h>
#include <hip/hip_bf16.h>
#include <math.h>

#define B_BATCH 2
#define N_PTS   8192
#define NPOINT_ 2048
#define NSAMPLE_ 32
#define IN_CH_  64
#define RADIUS2 0.04f
#define EPS_LN  1e-5f
#define TOK     (B_BATCH*NPOINT_*NSAMPLE_)   // 131072 tokens

typedef __bf16 v16bf __attribute__((ext_vector_type(16)));
typedef __bf16 v8bf  __attribute__((ext_vector_type(8)));
typedef float  v8f   __attribute__((ext_vector_type(8)));

__device__ __forceinline__ float gelu_exact(float x){
    return 0.5f * x * (1.0f + erff(x * 0.70710678118654752440f));
}

// ---------------------------------------------------------------------------
// Pack a row-major f32 weight W[K][N] into bf16 WMMA B-fragment order:
// tile t = kt*(N/16)+nt ; within tile: lane 0..31 x element 0..15 contiguous.
// lane -> n = nt*16 + (lane&15); element e -> k = kt*32 + (lane>>4)*8 + (e&7) + (e>>3)*16.
// ---------------------------------------------------------------------------
__global__ void pack_weight_bf16(const float* __restrict__ W, __bf16* __restrict__ P,
                                 int K, int N){
    int p = blockIdx.x * blockDim.x + threadIdx.x;
    if (p >= K * N) return;
    int t    = p >> 9;          // tile (512 elems)
    int r    = p & 511;
    int lane = r >> 4;
    int e    = r & 15;
    int ntn  = N >> 4;
    int kt   = t / ntn, nt = t % ntn;
    int n    = nt * 16 + (lane & 15);
    int k    = kt * 32 + (lane >> 4) * 8 + (e & 7) + (e >> 3) * 16;
    P[p] = (__bf16)W[k * N + n];
}

// ---------------------------------------------------------------------------
// Farthest point sampling: one workgroup per batch, 1024 threads x 8 points.
// Sequential 2048-step argmax loop; dist kept in registers.
// ---------------------------------------------------------------------------
__global__ __launch_bounds__(1024) void fps_kernel(const float* __restrict__ xyz,
                                                   int* __restrict__ fpsIdx){
    const int b = blockIdx.x;
    const float* X = xyz + (size_t)b * N_PTS * 3;
    const int t = threadIdx.x;
    const int PPT = N_PTS / 1024;
    float dist[PPT];
    #pragma unroll
    for (int i = 0; i < PPT; ++i) dist[i] = 1e10f;

    __shared__ float sVal[32];
    __shared__ int   sIdx[32];
    __shared__ int   sFar;

    int far = 0;
    for (int it = 0; it < NPOINT_; ++it){
        if (t == 0) fpsIdx[b * NPOINT_ + it] = far;
        float cx = X[far*3+0], cy = X[far*3+1], cz = X[far*3+2];
        float bestV = -1.0f; int bestI = 0;
        #pragma unroll
        for (int i = 0; i < PPT; ++i){
            int p = t + i * 1024;
            float dx = X[p*3+0]-cx, dy = X[p*3+1]-cy, dz = X[p*3+2]-cz;
            float d  = dx*dx + dy*dy + dz*dz;
            float nd = fminf(dist[i], d);
            dist[i] = nd;
            if (nd > bestV || (nd == bestV && p < bestI)) { bestV = nd; bestI = p; }
        }
        #pragma unroll
        for (int off = 16; off > 0; off >>= 1){
            float ov = __shfl_xor(bestV, off, 32);
            int   oi = __shfl_xor(bestI, off, 32);
            if (ov > bestV || (ov == bestV && oi < bestI)) { bestV = ov; bestI = oi; }
        }
        int wave = t >> 5, lane = t & 31;
        if (lane == 0){ sVal[wave] = bestV; sIdx[wave] = bestI; }
        __syncthreads();
        if (t == 0){
            float v = sVal[0]; int ix = sIdx[0];
            for (int w = 1; w < 32; ++w)
                if (sVal[w] > v || (sVal[w] == v && sIdx[w] < ix)) { v = sVal[w]; ix = sIdx[w]; }
            sFar = ix;
        }
        __syncthreads();
        far = sFar;
    }
}

__global__ void gather_newxyz(const float* __restrict__ xyz, const int* __restrict__ fpsIdx,
                              float* __restrict__ outXyz){
    int id = blockIdx.x * blockDim.x + threadIdx.x;
    if (id >= B_BATCH * NPOINT_) return;
    int b = id / NPOINT_;
    int fi = fpsIdx[id];
    const float* src = xyz + ((size_t)b * N_PTS + fi) * 3;
    outXyz[id*3+0] = src[0]; outXyz[id*3+1] = src[1]; outXyz[id*3+2] = src[2];
}

// ---------------------------------------------------------------------------
// Ball query: one wave32 per query; ballot + prefix-popcount keeps the first
// (smallest-index) 32 in-radius neighbors, matching top_k(-idx) semantics.
// ---------------------------------------------------------------------------
__global__ __launch_bounds__(256) void ball_query(const float* __restrict__ xyz,
                                                  const float* __restrict__ newXyz,
                                                  int* __restrict__ idxBuf){
    int q = blockIdx.x * (blockDim.x >> 5) + (threadIdx.x >> 5);
    int lane = threadIdx.x & 31;
    if (q >= B_BATCH * NPOINT_) return;
    int b = q / NPOINT_;
    const float* X = xyz + (size_t)b * N_PTS * 3;
    float cx = newXyz[q*3+0], cy = newXyz[q*3+1], cz = newXyz[q*3+2];
    int* out = idxBuf + (size_t)q * NSAMPLE_;
    int count = 0, first = 0;
    bool hasFirst = false;
    for (int base = 0; base < N_PTS; base += 32){
        int p = base + lane;
        float dx = X[p*3+0]-cx, dy = X[p*3+1]-cy, dz = X[p*3+2]-cz;
        bool pred = (dx*dx + dy*dy + dz*dz) <= RADIUS2;
        unsigned mask = __builtin_amdgcn_ballot_w32(pred);
        if (!hasFirst && mask){ first = base + __builtin_ctz(mask); hasFirst = true; }
        int before = __builtin_popcount(mask & ((1u << lane) - 1u));
        int pos = count + before;
        if (pred && pos < NSAMPLE_) out[pos] = p;
        count += __builtin_popcount(mask);
        if (count >= NSAMPLE_) break;
    }
    for (int j = count + lane; j < NSAMPLE_; j += 32) out[j] = first;
}

// ---------------------------------------------------------------------------
// Build X0 (bf16, [TOK][128]): ch 0..63 gathered points, ch 64..127 the
// gelu(LN(grouped_xyz @ enc_w + b)) positional encoding. One thread / token.
// ---------------------------------------------------------------------------
__global__ __launch_bounds__(128) void build_x0(const float* __restrict__ xyz,
                                                const float* __restrict__ points,
                                                const float* __restrict__ newXyz,
                                                const int* __restrict__ idxBuf,
                                                const float* __restrict__ encW,
                                                const float* __restrict__ encB,
                                                const float* __restrict__ encG,
                                                const float* __restrict__ encBeta,
                                                __bf16* __restrict__ X0){
    int t = blockIdx.x * blockDim.x + threadIdx.x;
    if (t >= TOK) return;
    int b   = t / (NPOINT_ * NSAMPLE_);
    int rem = t % (NPOINT_ * NSAMPLE_);
    int s   = rem / NSAMPLE_;
    int pidx = idxBuf[t];
    int q = b * NPOINT_ + s;
    const float* pp = xyz + ((size_t)b * N_PTS + pidx) * 3;
    float gx = pp[0] - newXyz[q*3+0];
    float gy = pp[1] - newXyz[q*3+1];
    float gz = pp[2] - newXyz[q*3+2];
    float v[64], sum = 0.f, sumsq = 0.f;
    #pragma unroll 8
    for (int c = 0; c < 64; ++c){
        float h = gx*encW[c] + gy*encW[64+c] + gz*encW[128+c] + encB[c];
        v[c] = h; sum += h; sumsq += h*h;
    }
    float mu  = sum * (1.f/64.f);
    float var = sumsq * (1.f/64.f) - mu*mu;
    float rs  = rsqrtf(var + EPS_LN);
    __bf16* dst = X0 + (size_t)t * 128;
    const float* ptf = points + ((size_t)b * N_PTS + pidx) * IN_CH_;
    #pragma unroll 8
    for (int c = 0; c < 64; ++c) dst[c] = (__bf16)ptf[c];
    #pragma unroll 8
    for (int c = 0; c < 64; ++c){
        float h = (v[c]-mu)*rs*encG[c] + encBeta[c];
        dst[64+c] = (__bf16)gelu_exact(h);
    }
}

// ---------------------------------------------------------------------------
// Fused inverted-residual block: out = gelu(LN(x@ew+eb))@pw + pb + x@sw.
// 256 threads (8 waves) per 16-token tile. The mid activation lives only in
// LDS: f32 copy (padded stride, bank-conflict-free) for the LayerNorm
// reductions, bf16 copy (row-major, 16B-aligned padded stride) so the
// projection GEMM's A-fragments are two ds_load_b128 per lane per k-step.
// All GEMMs are v_wmma_f32_16x16x32_bf16 with f32 accumulation.
// ---------------------------------------------------------------------------
template<int DIN, int DMID, int DOUT, bool OUT_BF16>
__global__ __launch_bounds__(256) void irb_block(
        const __bf16* __restrict__ Xin,
        const __bf16* __restrict__ Wew, const float* __restrict__ eb,
        const float* __restrict__ ng,  const float* __restrict__ nb,
        const __bf16* __restrict__ Wpw, const float* __restrict__ pb,
        const __bf16* __restrict__ Wsw,
        __bf16* __restrict__ XoutB, float* __restrict__ XoutF){
    constexpr int LDF = DMID + 2;   // f32 row stride (banks de-phased per row)
    constexpr int LDB = DMID + 8;   // bf16 row stride (+16B: keeps 16B align)
    __shared__ __align__(16) float  sh [16 * LDF];
    __shared__ __align__(16) __bf16 hbf[16 * LDB];
    const int wave  = threadIdx.x >> 5;
    const int lane  = threadIdx.x & 31;
    const int m     = lane & 15;
    const int khalf = lane >> 4;
    const int rowBase = blockIdx.x * 16;
    constexpr int NT_MID = DMID / 16;
    constexpr int TPW    = NT_MID / 8;    // n-tiles per wave in expand phase
    constexpr int KS_IN  = DIN / 32;
    constexpr int KS_MID = DMID / 32;
    constexpr int NT_OUT = DOUT / 16;

    const __bf16* arow = Xin + (size_t)(rowBase + m) * DIN + khalf * 8;

    // ---- Phase 1: expand GEMM  h = x @ ew  (+eb) -> LDS -------------------
    v8f acc[TPW];
    #pragma unroll
    for (int j = 0; j < TPW; ++j) { v8f z = {}; acc[j] = z; }

    #pragma unroll
    for (int ks = 0; ks < KS_IN; ++ks){
        const __bf16* ap = arow + ks * 32;
        v8bf lo = *(const v8bf*)(ap);
        v8bf hi = *(const v8bf*)(ap + 16);
        v16bf a = __builtin_shufflevector(lo, hi, 0,1,2,3,4,5,6,7,8,9,10,11,12,13,14,15);
        #pragma unroll
        for (int j = 0; j < TPW; ++j){
            int nt = wave * TPW + j;
            v16bf bf = *(const v16bf*)(Wew + ((size_t)(ks * NT_MID + nt) << 9) + lane * 16);
            acc[j] = __builtin_amdgcn_wmma_f32_16x16x32_bf16(false, a, false, bf,
                                                             (short)0, acc[j], false, false);
        }
    }
    #pragma unroll
    for (int j = 0; j < TPW; ++j){
        int ncol = (wave * TPW + j) * 16 + m;
        float bia = eb[ncol];
        #pragma unroll
        for (int v = 0; v < 8; ++v){
            int row = v + 8 * khalf;            // C/D layout: VGPR v <-> row
            sh[row * LDF + ncol] = acc[j][v] + bia;
        }
    }
    __syncthreads();

    // ---- Phase 2: row-wise LayerNorm + GELU -> bf16 LDS -------------------
    #pragma unroll
    for (int rr = 0; rr < 2; ++rr){
        int row = wave * 2 + rr;
        float sum = 0.f, sumsq = 0.f;
        for (int c = lane; c < DMID; c += 32){ float h = sh[row*LDF+c]; sum += h; sumsq += h*h; }
        #pragma unroll
        for (int off = 16; off > 0; off >>= 1){
            sum   += __shfl_xor(sum,   off, 32);
            sumsq += __shfl_xor(sumsq, off, 32);
        }
        float mu  = sum / (float)DMID;
        float var = sumsq / (float)DMID - mu*mu;
        float rs  = rsqrtf(var + EPS_LN);
        for (int c = lane; c < DMID; c += 32){
            float h = (sh[row*LDF+c] - mu) * rs * ng[c] + nb[c];
            hbf[row*LDB+c] = (__bf16)gelu_exact(h);
        }
    }
    __syncthreads();

    // ---- Phase 3: project + skip  out = h@pw + x@sw + pb ------------------
    if (wave < NT_OUT){
        v8f acc2 = {};
        #pragma unroll
        for (int ks = 0; ks < KS_MID; ++ks){
            int kb = ks * 32 + khalf * 8;       // A layout: e<8 -> kb+e, else kb+16+(e-8)
            v8bf lo = *(const v8bf*)(&hbf[m * LDB + kb]);
            v8bf hi = *(const v8bf*)(&hbf[m * LDB + kb + 16]);
            v16bf a = __builtin_shufflevector(lo, hi, 0,1,2,3,4,5,6,7,8,9,10,11,12,13,14,15);
            v16bf bf = *(const v16bf*)(Wpw + ((size_t)(ks * NT_OUT + wave) << 9) + lane * 16);
            acc2 = __builtin_amdgcn_wmma_f32_16x16x32_bf16(false, a, false, bf,
                                                           (short)0, acc2, false, false);
        }
        #pragma unroll
        for (int ks = 0; ks < KS_IN; ++ks){
            const __bf16* ap = arow + ks * 32;
            v8bf lo = *(const v8bf*)(ap);
            v8bf hi = *(const v8bf*)(ap + 16);
            v16bf a = __builtin_shufflevector(lo, hi, 0,1,2,3,4,5,6,7,8,9,10,11,12,13,14,15);
            v16bf bf = *(const v16bf*)(Wsw + ((size_t)(ks * NT_OUT + wave) << 9) + lane * 16);
            acc2 = __builtin_amdgcn_wmma_f32_16x16x32_bf16(false, a, false, bf,
                                                           (short)0, acc2, false, false);
        }
        int ncol = wave * 16 + m;
        float bia = pb[ncol];
        #pragma unroll
        for (int v = 0; v < 8; ++v){
            int row = v + 8 * khalf;
            float val = acc2[v] + bia;
            size_t o = (size_t)(rowBase + row) * DOUT + ncol;
            if (OUT_BF16) XoutB[o] = (__bf16)val; else XoutF[o] = val;
        }
    }
}

__global__ void maxpool_out(const float* __restrict__ X2, float* __restrict__ outPts){
    int id = blockIdx.x * blockDim.x + threadIdx.x;
    if (id >= B_BATCH * NPOINT_ * 128) return;
    int grp = id >> 7;
    int c   = id & 127;
    const float* p = X2 + (size_t)grp * NSAMPLE_ * 128 + c;
    float mv = p[0];
    #pragma unroll 8
    for (int k = 1; k < NSAMPLE_; ++k) mv = fmaxf(mv, p[k * 128]);
    outPts[id] = mv;
}

// ---------------------------------------------------------------------------
extern "C" void kernel_launch(void* const* d_in, const int* in_sizes, int n_in,
                              void* d_out, int out_size, void* d_ws, size_t ws_size,
                              hipStream_t stream){
    const float *xyz, *points, *encW, *encB, *encG, *encBeta;
    const float *ew0,*eb0,*ng0,*nb0,*pw0,*pb0,*sw0;
    const float *ew1,*eb1,*ng1,*nb1,*pw1,*pb1,*sw1;
    #define F(i) ((const float*)d_in[(i)])
    if (in_sizes[0] == B_BATCH * N_PTS * 3){
        // insertion-order flattening: xyz, points, enc*, block0 fields, block1 fields
        xyz=F(0); points=F(1); encW=F(2); encB=F(3); encG=F(4); encBeta=F(5);
        ew0=F(6); eb0=F(7); ng0=F(8); nb0=F(9); pw0=F(10); pb0=F(11); sw0=F(12);
        ew1=F(13); eb1=F(14); ng1=F(15); nb1=F(16); pw1=F(17); pb1=F(18); sw1=F(19);
    } else {
        // sorted pytree flattening: blocks (eb,ew,nb,ng,pb,pw,sw)x2, enc_b, enc_beta, enc_g, enc_w, points, xyz
        eb0=F(0); ew0=F(1); nb0=F(2); ng0=F(3); pb0=F(4); pw0=F(5); sw0=F(6);
        eb1=F(7); ew1=F(8); nb1=F(9); ng1=F(10); pb1=F(11); pw1=F(12); sw1=F(13);
        encB=F(14); encBeta=F(15); encG=F(16); encW=F(17); points=F(18); xyz=F(19);
    }
    #undef F

    char* ws = (char*)d_ws;
    size_t off = 0;
    auto alloc = [&](size_t bytes)->char*{
        char* p = ws + off;
        off += (bytes + 255) & ~(size_t)255;
        return p;
    };
    __bf16* Pew0 = (__bf16*)alloc((size_t)128*512*2);
    __bf16* Ppw0 = (__bf16*)alloc((size_t)512*64*2);
    __bf16* Psw0 = (__bf16*)alloc((size_t)128*64*2);
    __bf16* Pew1 = (__bf16*)alloc((size_t)64*256*2);
    __bf16* Ppw1 = (__bf16*)alloc((size_t)256*128*2);
    __bf16* Psw1 = (__bf16*)alloc((size_t)64*128*2);
    int*    fpsIdx = (int*)alloc((size_t)B_BATCH*NPOINT_*4);
    int*    idxBuf = (int*)alloc((size_t)B_BATCH*NPOINT_*NSAMPLE_*4);
    __bf16* X0 = (__bf16*)alloc((size_t)TOK*128*2);
    __bf16* X1 = (__bf16*)alloc((size_t)TOK*64*2);
    float*  X2 = (float*) alloc((size_t)TOK*128*4);

    float* outXyz = (float*)d_out;                       // (2,2048,3)
    float* outPts = (float*)d_out + B_BATCH*NPOINT_*3;   // (2,2048,128)

    // 1) pack weights to WMMA B-fragment layout (bf16)
    pack_weight_bf16<<<(128*512+255)/256,256,0,stream>>>(ew0, Pew0, 128, 512);
    pack_weight_bf16<<<(512*64 +255)/256,256,0,stream>>>(pw0, Ppw0, 512, 64);
    pack_weight_bf16<<<(128*64 +255)/256,256,0,stream>>>(sw0, Psw0, 128, 64);
    pack_weight_bf16<<<(64*256 +255)/256,256,0,stream>>>(ew1, Pew1, 64, 256);
    pack_weight_bf16<<<(256*128+255)/256,256,0,stream>>>(pw1, Ppw1, 256, 128);
    pack_weight_bf16<<<(64*128 +255)/256,256,0,stream>>>(sw1, Psw1, 64, 128);

    // 2) FPS + gather centers
    fps_kernel<<<B_BATCH, 1024, 0, stream>>>(xyz, fpsIdx);
    gather_newxyz<<<(B_BATCH*NPOINT_+255)/256,256,0,stream>>>(xyz, fpsIdx, outXyz);

    // 3) ball query (one wave per query)
    ball_query<<<(B_BATCH*NPOINT_+7)/8, 256, 0, stream>>>(xyz, outXyz, idxBuf);

    // 4) gather + positional encoder -> X0 bf16 [TOK][128]
    build_x0<<<(TOK+127)/128,128,0,stream>>>(xyz, points, outXyz, idxBuf,
                                             encW, encB, encG, encBeta, X0);

    // 5) fused inverted-residual blocks (WMMA bf16)
    irb_block<128,512,64,true ><<<TOK/16,256,0,stream>>>(X0, Pew0, eb0, ng0, nb0,
                                                         Ppw0, pb0, Psw0, X1, nullptr);
    irb_block<64,256,128,false><<<TOK/16,256,0,stream>>>(X1, Pew1, eb1, ng1, nb1,
                                                         Ppw1, pb1, Psw1, nullptr, X2);

    // 6) max over the 32 samples -> new_points
    maxpool_out<<<(B_BATCH*NPOINT_*128+255)/256,256,0,stream>>>(X2, outPts);
}
